// NequIP_76613626626648
// MI455X (gfx1250) — matrix operand order repaired and checked by reading, not compiled, not run on hardware
//
#include <hip/hip_runtime.h>
#include <hip/hip_bf16.h>

typedef __attribute__((ext_vector_type(16))) _Float16 v16h;
typedef __attribute__((ext_vector_type(8)))  float    v8f;

#define NCH 64            // LATENT channels
#define NB 8              // radial basis
#define RCUT 5.0f
#define PI_F 3.14159265358979f
#define NODES_PER_GRAPH 200
#define FLATD 576         // 64*(1+3+5)
#define EDGE_TILES_PER_BLOCK 4

__device__ __forceinline__ float fast_rcp(float x) { return __builtin_amdgcn_rcpf(x); }
__device__ __forceinline__ float silu_f(float x) { return x * __builtin_amdgcn_rcpf(1.f + __expf(-x)); }
__device__ __forceinline__ float sigm_f(float x) { return __builtin_amdgcn_rcpf(1.f + __expf(-x)); }

// A fragment: 16xK tile, row-major in LDS with leading dim ldk.
// lane<16: row=lane, K = k0..k0+15 ; lane>=16: row=lane-16, K = k0+16..k0+31
__device__ __forceinline__ v16h load_a_frag(const _Float16* tile, int ldk, int k0, int lane) {
    int m = lane & 15;
    int kb = k0 + ((lane & 16) ? 16 : 0);
    v16h a;
#pragma unroll
    for (int i = 0; i < 16; ++i) a[i] = tile[m * ldk + kb + i];
    return a;
}

// B fragment: 32x16 tile of [K x N] row-major f16 (leading dim ldn), at (k0, n0).
// lane l holds column n = n0 + (l & 15), rows k = k0 + 16*(l/16) + 0..15.
__device__ __forceinline__ v16h load_b_frag(const _Float16* w, int ldn, int k0, int n0, int lane) {
    const _Float16* row = w + (k0 + ((lane & 16) ? 16 : 0)) * ldn + n0 + (lane & 15);
    v16h b;
#pragma unroll
    for (int i = 0; i < 16; ++i) b[i] = row[i * ldn];
    return b;
}

__device__ __forceinline__ v8f wmma_f16(v16h a, v16h b, v8f c) {
    return __builtin_amdgcn_wmma_f32_16x16x32_f16(false, a, false, b, (short)0, c, false, false);
}

// ---------------------------------------------------------------- kernel 1
__global__ __launch_bounds__(256) void embed_kernel(const int* __restrict__ species,
                                                    const float* __restrict__ node_embed,
                                                    float* __restrict__ h, int nTot) {
    int i = blockIdx.x * 256 + threadIdx.x;
    if (i < nTot) h[i] = node_embed[species[i >> 6] * NCH + (i & 63)];
}

// ---------------------------------------------------------------- kernel 2 (edges)
// 8 waves/block, 16 edges/wave, EDGE_TILES_PER_BLOCK tiles of 128 edges per block.
// Radial MLP via WMMA, fused message + scatter.
__global__ __launch_bounds__(256) void edge_kernel(
    const float* __restrict__ pos, const int* __restrict__ senders,
    const int* __restrict__ receivers, const float* __restrict__ h,
    const float* __restrict__ rw1, const float* __restrict__ rw2,
    const float* __restrict__ rw3,
    float* __restrict__ a0, float* __restrict__ a1, float* __restrict__ a2,
    int nEdges) {
    __shared__ _Float16 w1s[32 * 64];       // K padded 8->32
    __shared__ _Float16 w2s[64 * 64];
    __shared__ _Float16 w3s[64 * 192];
    __shared__ _Float16 stage[8][16 * 64];  // per-wave staging tile
    __shared__ float    shv[8][16][8];      // sh1(3)+sh2(5) per edge
    __shared__ int      sidx[8][16];
    __shared__ int      ridx[8][16];

    const int tid = threadIdx.x;
    for (int i = tid; i < 32 * 64; i += 256) {
        int k = i >> 6, n = i & 63;
        w1s[i] = (k < NB) ? (_Float16)rw1[k * 64 + n] : (_Float16)0.f;
    }
    for (int i = tid; i < 64 * 64; i += 256) w2s[i] = (_Float16)rw2[i];
    for (int i = tid; i < 64 * 192; i += 256) w3s[i] = (_Float16)rw3[i];

    const int wave = tid >> 5, lane = tid & 31;
    _Float16* buf = stage[wave];
    const int nn = lane & 15, mb = (lane < 16) ? 0 : 8;
    const int base = blockIdx.x * (128 * EDGE_TILES_PER_BLOCK);

    for (int it = 0; it < EDGE_TILES_PER_BLOCK; ++it) {
        const int e0 = base + it * 128 + wave * 16;

        // ---- phase A: per-edge geometry, SH, Bessel basis -> buf[16x32]
        if (lane < 16) {
            int e = e0 + lane;
            bool valid = e < nEdges;
            int s = valid ? senders[e] : 0;
            int rc = valid ? receivers[e] : 0;
            sidx[wave][lane] = s;
            ridx[wave][lane] = rc;
            float vx = pos[rc * 3 + 0] - pos[s * 3 + 0];
            float vy = pos[rc * 3 + 1] - pos[s * 3 + 1];
            float vz = pos[rc * 3 + 2] - pos[s * 3 + 2];
            float r = sqrtf(vx * vx + vy * vy + vz * vz);
            float rinv = fast_rcp(r + 1e-9f);
            float ux = vx * rinv, uy = vy * rinv, uz = vz * rinv;
            const float s3 = 1.7320508075688772f, s15 = 3.872983346207417f, s5 = 2.23606797749979f;
            shv[wave][lane][0] = s3 * ux;
            shv[wave][lane][1] = s3 * uy;
            shv[wave][lane][2] = s3 * uz;
            shv[wave][lane][3] = s15 * ux * uy;
            shv[wave][lane][4] = s15 * uy * uz;
            shv[wave][lane][5] = 0.5f * s5 * (3.f * uz * uz - 1.f);
            shv[wave][lane][6] = s15 * ux * uz;
            shv[wave][lane][7] = 0.5f * s15 * (ux * ux - uy * uy);
            float env = 0.5f * (__cosf(PI_F * fminf(r * (1.f / RCUT), 1.f)) + 1.f);
            float pref = valid ? (sqrtf(2.f / RCUT) * rinv * env) : 0.f;
#pragma unroll
            for (int n = 0; n < 32; ++n) {
                float v = (n < NB) ? pref * __sinf((float)(n + 1) * (PI_F / RCUT) * r) : 0.f;
                buf[lane * 32 + n] = (_Float16)v;
            }
        }
        __syncthreads();

        // ---- GEMM1: hid = silu(rb @ w1)   [16x32]@[32x64]
        v16h arb = load_a_frag(buf, 32, 0, lane);
        v8f hidT[4];
#pragma unroll
        for (int t = 0; t < 4; ++t) {
            v8f c = {};
            c = wmma_f16(arb, load_b_frag(w1s, 64, 0, t * 16, lane), c);
#pragma unroll
            for (int v = 0; v < 8; ++v) hidT[t][v] = silu_f(c[v]);
        }
        __syncthreads();
#pragma unroll
        for (int t = 0; t < 4; ++t)
#pragma unroll
            for (int v = 0; v < 8; ++v) buf[(mb + v) * 64 + t * 16 + nn] = (_Float16)hidT[t][v];
        __syncthreads();

        // ---- GEMM2: hid2 = silu(hid @ w2)  [16x64]@[64x64]
        v16h af0 = load_a_frag(buf, 64, 0, lane);
        v16h af1 = load_a_frag(buf, 64, 32, lane);
        v8f h2T[4];
#pragma unroll
        for (int t = 0; t < 4; ++t) {
            v8f c = {};
            c = wmma_f16(af0, load_b_frag(w2s, 64, 0, t * 16, lane), c);
            c = wmma_f16(af1, load_b_frag(w2s, 64, 32, t * 16, lane), c);
#pragma unroll
            for (int v = 0; v < 8; ++v) h2T[t][v] = silu_f(c[v]);
        }
        __syncthreads();
#pragma unroll
        for (int t = 0; t < 4; ++t)
#pragma unroll
            for (int v = 0; v < 8; ++v) buf[(mb + v) * 64 + t * 16 + nn] = (_Float16)h2T[t][v];
        __syncthreads();

        // ---- GEMM3: wts = hid2 @ w3  [16x64]@[64x192], fused message + scatter
        v16h c0 = load_a_frag(buf, 64, 0, lane);
        v16h c1 = load_a_frag(buf, 64, 32, lane);
        __syncthreads();
        // stage h[senders] tile into buf (f16)
        for (int idx = lane; idx < 16 * 64; idx += 32) {
            int m = idx >> 6, ch = idx & 63;
            buf[idx] = (_Float16)h[sidx[wave][m] * NCH + ch];
        }
        __syncthreads();

        const float inv = 0.25f;  // 1/sqrt(AVG_NUM_NEIGHBORS)
#pragma unroll
        for (int t = 0; t < 12; ++t) {
            v8f c = {};
            c = wmma_f16(c0, load_b_frag(w3s, 192, 0, t * 16, lane), c);
            c = wmma_f16(c1, load_b_frag(w3s, 192, 32, t * 16, lane), c);
            int col = t * 16 + nn;  // 0..191
#pragma unroll
            for (int v = 0; v < 8; ++v) {
                int m = mb + v;
                int e = e0 + m;
                if (e >= nEdges) continue;
                float hsv = (float)buf[m * 64 + (col & 63)];
                float msg = c[v] * hsv * inv;
                int rcv = ridx[wave][m];
                if (col < 64) {
                    atomicAdd(&a0[rcv * NCH + col], msg);
                } else if (col < 128) {
                    int ch = col - 64;
#pragma unroll
                    for (int j = 0; j < 3; ++j)
                        atomicAdd(&a1[(rcv * NCH + ch) * 3 + j], msg * shv[wave][m][j]);
                } else {
                    int ch = col - 128;
#pragma unroll
                    for (int j = 0; j < 5; ++j)
                        atomicAdd(&a2[(rcv * NCH + ch) * 5 + j], msg * shv[wave][m][3 + j]);
                }
            }
        }
        __syncthreads();
    }
}

// ---------------------------------------------------------------- kernel 3 (nodes)
__global__ __launch_bounds__(256) void node_kernel(
    const float* __restrict__ a0, const float* __restrict__ a1,
    const float* __restrict__ a2, const float* __restrict__ h,
    const int* __restrict__ species,
    const float* __restrict__ lin_w0, const float* __restrict__ lin_w1,
    const float* __restrict__ lin_w2, const float* __restrict__ skip_w,
    const float* __restrict__ gate_w, const float* __restrict__ focus_w,
    const float* __restrict__ focus_b,
    float* __restrict__ focus_logits, float* __restrict__ focus_emb, int nNodes) {
    __shared__ _Float16 lw0[64 * 64], lw1[64 * 64], lw2[64 * 64];
    __shared__ _Float16 gw[64 * 128];
    __shared__ _Float16 stage[8][16 * 64];
    __shared__ float    facc[8][16];
    __shared__ int      specArr[8][16];

    const int tid = threadIdx.x;
    for (int i = tid; i < 64 * 64; i += 256) {
        lw0[i] = (_Float16)lin_w0[i];
        lw1[i] = (_Float16)lin_w1[i];
        lw2[i] = (_Float16)lin_w2[i];
    }
    for (int i = tid; i < 64 * 128; i += 256) gw[i] = (_Float16)gate_w[i];

    const int wave = tid >> 5, lane = tid & 31;
    const int n0 = blockIdx.x * 128 + wave * 16;
    _Float16* buf = stage[wave];
    const int nn = lane & 15, mb = (lane < 16) ? 0 : 8;

    if (lane < 16) {
        facc[wave][lane] = 0.f;
        int node = n0 + lane;
        specArr[wave][lane] = (node < nNodes) ? species[node] : 0;
    }

    // stage a0 tile
    for (int idx = lane; idx < 16 * 64; idx += 32) {
        int m = idx >> 6, ch = idx & 63;
        int node = n0 + m;
        buf[idx] = (node < nNodes) ? (_Float16)a0[node * NCH + ch] : (_Float16)0.f;
    }
    __syncthreads();
    v16h af0 = load_a_frag(buf, 64, 0, lane);
    v16h af1 = load_a_frag(buf, 64, 32, lane);
    __syncthreads();
    // stage node feature h tile (for skip connection)
    for (int idx = lane; idx < 16 * 64; idx += 32) {
        int m = idx >> 6, ch = idx & 63;
        int node = n0 + m;
        buf[idx] = (node < nNodes) ? (_Float16)h[node * NCH + ch] : (_Float16)0.f;
    }
    __syncthreads();

    float accRow[8];
#pragma unroll
    for (int v = 0; v < 8; ++v) accRow[v] = 0.f;

    // s = a0 @ lin_w0 + skip;  keep s tiles, emit s_out contributions
    v8f sreg[4];
#pragma unroll
    for (int t = 0; t < 4; ++t) {
        v8f c = {};
        c = wmma_f16(af0, load_b_frag(lw0, 64, 0, t * 16, lane), c);
        c = wmma_f16(af1, load_b_frag(lw0, 64, 32, t * 16, lane), c);
        int d = t * 16 + nn;
        for (int v = 0; v < 8; ++v) {
            int m = mb + v;
            int node = n0 + m;
            float sk = 0.f;
            if (node < nNodes) {
                const float* skw = skip_w + specArr[wave][m] * 4096 + d;
                for (int ch = 0; ch < 64; ++ch) sk += (float)buf[m * 64 + ch] * skw[ch * 64];
            }
            sreg[t][v] = c[v] + sk;
        }
    }
    // s_out contributions to focus / focus_emb
#pragma unroll
    for (int t = 0; t < 4; ++t) {
        int d = t * 16 + nn;
        float fw = focus_w[d];
        for (int v = 0; v < 8; ++v) {
            float so = silu_f(sreg[t][v]);
            accRow[v] += so * fw;
            int node = n0 + mb + v;
            if (node < nNodes && (node % NODES_PER_GRAPH) == 0)
                focus_emb[(node / NODES_PER_GRAPH) * FLATD + d] = so;
        }
    }
    __syncthreads();
    // stage s (pre-activation) for gate GEMM
#pragma unroll
    for (int t = 0; t < 4; ++t)
#pragma unroll
        for (int v = 0; v < 8; ++v) buf[(mb + v) * 64 + t * 16 + nn] = (_Float16)sreg[t][v];
    __syncthreads();
    af0 = load_a_frag(buf, 64, 0, lane);
    af1 = load_a_frag(buf, 64, 32, lane);

    // gates = sigmoid(s @ gate_w)  -> 8 tiles kept in registers
    v8f g[8];
#pragma unroll
    for (int t = 0; t < 8; ++t) {
        v8f c = {};
        c = wmma_f16(af0, load_b_frag(gw, 128, 0, t * 16, lane), c);
        c = wmma_f16(af1, load_b_frag(gw, 128, 32, t * 16, lane), c);
#pragma unroll
        for (int v = 0; v < 8; ++v) g[t][v] = sigm_f(c[v]);
    }

    // v_out: 3 components through lin_w1, gated by g[0..3]
    for (int j = 0; j < 3; ++j) {
        __syncthreads();
        for (int idx = lane; idx < 16 * 64; idx += 32) {
            int m = idx >> 6, ch = idx & 63;
            int node = n0 + m;
            buf[idx] = (node < nNodes) ? (_Float16)a1[(node * NCH + ch) * 3 + j] : (_Float16)0.f;
        }
        __syncthreads();
        v16h b0 = load_a_frag(buf, 64, 0, lane);
        v16h b1 = load_a_frag(buf, 64, 32, lane);
#pragma unroll
        for (int t = 0; t < 4; ++t) {
            v8f c = {};
            c = wmma_f16(b0, load_b_frag(lw1, 64, 0, t * 16, lane), c);
            c = wmma_f16(b1, load_b_frag(lw1, 64, 32, t * 16, lane), c);
            int d = t * 16 + nn;
            int idx2 = 64 + d * 3 + j;
            float fw = focus_w[idx2];
            for (int v = 0; v < 8; ++v) {
                float vo = c[v] * g[t][v];
                accRow[v] += vo * fw;
                int node = n0 + mb + v;
                if (node < nNodes && (node % NODES_PER_GRAPH) == 0)
                    focus_emb[(node / NODES_PER_GRAPH) * FLATD + idx2] = vo;
            }
        }
    }

    // t_out: 5 components through lin_w2, gated by g[4..7]
    for (int j = 0; j < 5; ++j) {
        __syncthreads();
        for (int idx = lane; idx < 16 * 64; idx += 32) {
            int m = idx >> 6, ch = idx & 63;
            int node = n0 + m;
            buf[idx] = (node < nNodes) ? (_Float16)a2[(node * NCH + ch) * 5 + j] : (_Float16)0.f;
        }
        __syncthreads();
        v16h b0 = load_a_frag(buf, 64, 0, lane);
        v16h b1 = load_a_frag(buf, 64, 32, lane);
#pragma unroll
        for (int t = 0; t < 4; ++t) {
            v8f c = {};
            c = wmma_f16(b0, load_b_frag(lw2, 64, 0, t * 16, lane), c);
            c = wmma_f16(b1, load_b_frag(lw2, 64, 32, t * 16, lane), c);
            int d = t * 16 + nn;
            int idx2 = 64 + 192 + d * 5 + j;
            float fw = focus_w[idx2];
            for (int v = 0; v < 8; ++v) {
                float to = c[v] * g[4 + t][v];
                accRow[v] += to * fw;
                int node = n0 + mb + v;
                if (node < nNodes && (node % NODES_PER_GRAPH) == 0)
                    focus_emb[(node / NODES_PER_GRAPH) * FLATD + idx2] = to;
            }
        }
    }

    // focus logit reduction across lanes via LDS atomics
#pragma unroll
    for (int v = 0; v < 8; ++v) atomicAdd(&facc[wave][mb + v], accRow[v]);
    __syncthreads();
    if (lane < 16) {
        int node = n0 + lane;
        if (node < nNodes) focus_logits[node] = facc[wave][lane] + focus_b[0];
    }
}

// ---------------------------------------------------------------- kernel 4 (graphs)
__global__ __launch_bounds__(256) void graph_kernel(
    const float* __restrict__ focus_emb, const float* __restrict__ species_embed,
    const int* __restrict__ target_species, const float* __restrict__ species_w,
    const float* __restrict__ species_b, const float* __restrict__ pos_w,
    const float* __restrict__ pos_b, float* __restrict__ species_logits,
    float* __restrict__ pos_out) {
    int gidx = blockIdx.x;
    int tid = threadIdx.x;
    __shared__ float pin[640];
    for (int i = tid; i < FLATD; i += 256) pin[i] = focus_emb[gidx * FLATD + i];
    int ts = target_species[gidx];
    for (int i = tid; i < 64; i += 256) pin[FLATD + i] = species_embed[ts * NCH + i];
    __syncthreads();
    if (tid < 5) {
        float acc = species_b[tid];
        for (int k = 0; k < FLATD; ++k) acc += pin[k] * species_w[k * 5 + tid];
        species_logits[gidx * 5 + tid] = acc;
    }
    for (int j = tid; j < 1600; j += 256) {
        float acc = pos_b[j];
        for (int k = 0; k < 640; ++k) acc += pin[k] * pos_w[k * 1600 + j];
        pos_out[gidx * 1600 + j] = acc;
    }
}

// ---------------------------------------------------------------- launch
extern "C" void kernel_launch(void* const* d_in, const int* in_sizes, int n_in,
                              void* d_out, int out_size, void* d_ws, size_t ws_size,
                              hipStream_t stream) {
    const float* positions     = (const float*)d_in[0];
    const int*   species       = (const int*)d_in[1];
    const int*   senders       = (const int*)d_in[2];
    const int*   receivers     = (const int*)d_in[3];
    const int*   target_species= (const int*)d_in[5];
    const float* species_embed = (const float*)d_in[6];
    const float* node_embed    = (const float*)d_in[7];
    const float* radial_w1     = (const float*)d_in[8];
    const float* radial_w2     = (const float*)d_in[9];
    const float* radial_w3     = (const float*)d_in[10];
    const float* lin_w0        = (const float*)d_in[11];
    const float* lin_w1        = (const float*)d_in[12];
    const float* lin_w2        = (const float*)d_in[13];
    const float* skip_w        = (const float*)d_in[14];
    const float* gate_w        = (const float*)d_in[15];
    const float* focus_w       = (const float*)d_in[16];
    const float* focus_b       = (const float*)d_in[17];
    const float* species_w     = (const float*)d_in[18];
    const float* species_b     = (const float*)d_in[19];
    const float* pos_w         = (const float*)d_in[20];
    const float* pos_b         = (const float*)d_in[21];

    const int nNodes = in_sizes[1];
    const int nEdges = in_sizes[2];
    const int nGraphs = in_sizes[4];

    float* ws = (float*)d_ws;
    float* hbuf = ws;                          size_t off = (size_t)nNodes * NCH;
    float* a0   = ws + off;                    off += (size_t)nNodes * NCH;
    float* a1   = ws + off;                    off += (size_t)nNodes * NCH * 3;
    float* a2   = ws + off;                    off += (size_t)nNodes * NCH * 5;
    float* femb = ws + off;                    off += (size_t)nGraphs * FLATD;

    float* focus_logits   = (float*)d_out;
    float* species_logits = focus_logits + nNodes;
    float* pos_out        = species_logits + (size_t)nGraphs * 5;

    // node embeddings
    int tot = nNodes * NCH;
    embed_kernel<<<(tot + 255) / 256, 256, 0, stream>>>(species, node_embed, hbuf, tot);

    // zero aggregation buffers (a0,a1,a2 contiguous)
    hipMemsetAsync(a0, 0, (size_t)nNodes * NCH * 9 * sizeof(float), stream);

    // edge message passing (WMMA radial MLP + fused scatter)
    const int edgesPerBlock = 128 * EDGE_TILES_PER_BLOCK;
    edge_kernel<<<(nEdges + edgesPerBlock - 1) / edgesPerBlock, 256, 0, stream>>>(
        positions, senders, receivers, hbuf, radial_w1, radial_w2, radial_w3,
        a0, a1, a2, nEdges);

    // node update (WMMA linear layers + gating + focus readout)
    node_kernel<<<(nNodes + 127) / 128, 256, 0, stream>>>(
        a0, a1, a2, hbuf, species, lin_w0, lin_w1, lin_w2, skip_w, gate_w,
        focus_w, focus_b, focus_logits, femb, nNodes);

    // graph-level heads
    graph_kernel<<<nGraphs, 256, 0, stream>>>(
        femb, species_embed, target_species, species_w, species_b,
        pos_w, pos_b, species_logits, pos_out);
}
